// LSNN_46823733461843
// MI455X (gfx1250) — matrix operand order repaired and checked by reading, not compiled, not run on hardware
//
#include <hip/hip_runtime.h>

typedef __attribute__((ext_vector_type(16))) __bf16 v16bf;
typedef __attribute__((ext_vector_type(8)))  float  v8f;

#define T_STEPS 500
#define B_SZ    128
#define NI_SZ   256
#define N_SZ    256
#define THR_C   0.03f
#define BETA_C  1.8f
#define DECAY_C 0.95122945f   /* exp(-1/20) in f32 */

#define NX_ELEMS (B_SZ * T_STEPS * NI_SZ)   /* 16,384,000 */
#define NW_ELEMS (NI_SZ * N_SZ)             /* 65,536 */

// ---- bf16 helpers via explicit RNE bit manipulation ----
__device__ __forceinline__ unsigned short f2bfbits(float f) {
  unsigned u = __float_as_uint(f);
  unsigned r = u + 0x7FFFu + ((u >> 16) & 1u);
  return (unsigned short)(r >> 16);
}
__device__ __forceinline__ float bfbits2f(unsigned short s) {
  return __uint_as_float(((unsigned)s) << 16);
}
__device__ __forceinline__ __bf16 bits2bf(unsigned short s) {
  __bf16 h; __builtin_memcpy(&h, &s, 2); return h;
}

// Build a v16bf fragment from two contiguous 16-byte runs (8 bf16 each).
__device__ __forceinline__ v16bf ldfrag2(const unsigned short* p0,
                                         const unsigned short* p1) {
  union { v16bf v; uint4 q[2]; } u;
  u.q[0] = *(const uint4*)p0;
  u.q[1] = *(const uint4*)p1;
  return u.v;
}
// 8-float (32B) vector load/store for fragment-layout f32 data.
__device__ __forceinline__ v8f ldf8(const float* p) {
  union { v8f v; uint4 q[2]; } u;
  u.q[0] = *(const uint4*)p;
  u.q[1] = *(const uint4*)(p + 4);
  return u.v;
}
__device__ __forceinline__ void stf8(float* p, v8f v) {
  union { v8f v; uint4 q[2]; } u; u.v = v;
  *(uint4*)p       = u.q[0];
  *(uint4*)(p + 4) = u.q[1];
}

// =======================================================================
// Pre-pack 0: split x into bf16 hi/lo planes (same [B][T][NI] layout).
// =======================================================================
__global__ __launch_bounds__(256) void lsnn_pack_x(
    const float* __restrict__ x,
    unsigned short* __restrict__ xh, unsigned short* __restrict__ xl) {
  int stride = gridDim.x * blockDim.x;
  for (int i = blockIdx.x * blockDim.x + threadIdx.x; i < NX_ELEMS; i += stride) {
    float f = x[i];
    unsigned short hb = f2bfbits(f);
    xh[i] = hb;
    xl[i] = f2bfbits(f - bfbits2f(hb));
  }
}

// =======================================================================
// Pre-pack 1: split w_in into bf16 hi/lo planes, TRANSPOSED to [N][NI]
// so B-fragments are contiguous runs along K.
// =======================================================================
__global__ __launch_bounds__(256) void lsnn_pack_w(
    const float* __restrict__ w_in,
    unsigned short* __restrict__ whT, unsigned short* __restrict__ wlT) {
  int i = blockIdx.x * blockDim.x + threadIdx.x;   // i = k*N + n
  if (i < NW_ELEMS) {
    int k = i >> 8, n = i & 255;
    float f = w_in[i];
    unsigned short hb = f2bfbits(f);
    whT[n * NI_SZ + k] = hb;
    wlT[n * NI_SZ + k] = f2bfbits(f - bfbits2f(hb));
  }
}

// =======================================================================
// Phase 1: input currents for one time step per wave-tile:
//   tile = 16 batch rows x 16 neurons at fixed t  (matches the scan tile).
// Output stored in NATIVE C/D fragment layout:
//   xwf[((t*8 + btile)*16 + ntile)*32 + lane][r]  (8 f32 per lane, contiguous)
// 3 independent accumulators (hh / lh / hl) -> WMMA chain length 8.
// =======================================================================
__global__ __launch_bounds__(128) void lsnn_xw_gemm(
    const unsigned short* __restrict__ xh, const unsigned short* __restrict__ xl,
    const unsigned short* __restrict__ whT, const unsigned short* __restrict__ wlT,
    float* __restrict__ xwf) {
  const int lane  = threadIdx.x & 31;
  const int gw    = blockIdx.x * 4 + (threadIdx.x >> 5);  // 64000 wave-tiles
  const int ntile = gw & 15;
  const int btile = (gw >> 4) & 7;
  const int t     = gw >> 7;
  const int lh    = (lane < 16) ? 0 : 1;
  const int ln    = lane & 15;

  const int    b    = btile * 16 + ln;                       // A row (this lane)
  const size_t arow = ((size_t)b * T_STEPS + t) * NI_SZ;
  const int    n    = ntile * 16 + ln;
  const size_t brow = (size_t)n * NI_SZ;                     // B row (transposed W)

  v8f accA = {0.f,0.f,0.f,0.f,0.f,0.f,0.f,0.f};
  v8f accB = accA, accC = accA;
#pragma unroll
  for (int kc = 0; kc < 8; ++kc) {
    const int offA0 = kc * 32 + lh * 8;        // A: K runs {..+7} and {+16..+23}
    const int offA1 = offA0 + 16;
    const int offB  = kc * 32 + lh * 8;        // B: K runs {..+7} and {+16..+23}
    v16bf ah = ldfrag2(xh + arow + offA0, xh + arow + offA1);
    v16bf al = ldfrag2(xl + arow + offA0, xl + arow + offA1);
    v16bf bh = ldfrag2(whT + brow + offB, whT + brow + offB + 16);
    v16bf bl = ldfrag2(wlT + brow + offB, wlT + brow + offB + 16);
    accA = __builtin_amdgcn_wmma_f32_16x16x32_bf16(false, ah, false, bh, (short)0, accA, false, false);
    accB = __builtin_amdgcn_wmma_f32_16x16x32_bf16(false, al, false, bh, (short)0, accB, false, false);
    accC = __builtin_amdgcn_wmma_f32_16x16x32_bf16(false, ah, false, bl, (short)0, accC, false, false);
  }
  v8f acc = (accA + accB) + accC;
  stf8(xwf + (size_t)gw * 256 + lane * 8, acc);   // 2x global_store_b128
}

// =======================================================================
// Phase 2: sequential scan. 8 blocks x 16 waves; block owns 16 batch rows,
// wave owns one 16-col N-tile. w_rec fragments (hi/lo bf16, diag masked)
// stay in registers for all 500 steps; z ping-pongs through LDS as bf16
// (conflict-free ds b128 fragment reads). Input currents arrive as two
// b128 loads in fragment layout. 4 accumulators -> WMMA chain length 4.
// =======================================================================
__global__ __launch_bounds__(512) void lsnn_scan(
    const float* __restrict__ w_rec, const float* __restrict__ xwf,
    const float* __restrict__ z0,   const float* __restrict__ v0,
    const float* __restrict__ a0,   const float* __restrict__ lsd0,
    float* __restrict__ out) {
  __shared__ __align__(16) unsigned short zbuf[2][16][264]; // 528B rows: 16B-aligned, bank-stride 4

  const int lane = threadIdx.x & 31;
  const int wave = threadIdx.x >> 5;
  const int lh   = (lane < 16) ? 0 : 1;
  const int ln   = lane & 15;
  const int n0   = wave * 16;
  const int b0   = blockIdx.x * 16;
  const int n    = n0 + ln;

  // --- one-time: w_rec fragments (masked diagonal), bf16 hi/lo split ---
  v16bf bh[8], bl[8];
#pragma unroll
  for (int kc = 0; kc < 8; ++kc) {
#pragma unroll
    for (int e = 0; e < 16; ++e) {
      int k = kc * 32 + lh * 16 + e;           // B layout
      float wv = w_rec[(size_t)k * N_SZ + n];
      if (k == n) wv = 0.f;                    // 1 - eye mask
      unsigned short hbit = f2bfbits(wv);
      bh[kc][e] = bits2bf(hbit);
      bl[kc][e] = bits2bf(f2bfbits(wv - bfbits2f(hbit)));
    }
  }

  // --- initial state (per-lane registers) + seed z into LDS buffer 0 ---
  float vreg[8], areg[8], lreg[8], zreg[8];
#pragma unroll
  for (int r = 0; r < 8; ++r) {
    int m = r + lh * 8;                        // C/D layout row
    size_t idx = (size_t)(b0 + m) * N_SZ + n;
    vreg[r] = v0[idx];
    areg[r] = a0[idx];
    lreg[r] = lsd0[idx];
    zreg[r] = z0[idx];
    zbuf[0][m][n] = f2bfbits(zreg[r]);
  }

  const int am = ln;                           // A layout: M = lane&15
  const size_t plane = (size_t)B_SZ * N_SZ;
  const size_t tvar  = (size_t)T_STEPS * plane;
  // this wave's slot in the fragment-layout current buffer
  const float* xwf_my = xwf + ((size_t)blockIdx.x * 16 + wave) * 256 + lane * 8;
  const size_t xwf_tstride = (size_t)8 * 16 * 256;   // one t step

  for (int t = 0; t < T_STEPS; ++t) {
    __syncthreads();                           // one barrier/step (double-buffered z)
    const int cur = t & 1, nxt = cur ^ 1;

    const float* xwp = xwf_my + (size_t)t * xwf_tstride;
    if (t + 1 < T_STEPS)                       // hide next step's current loads
      __builtin_prefetch(xwp + xwf_tstride, 0, 1);

    // recurrent current: R = Z @ Wrec  (Z exact in bf16; W = hi + lo)
    v8f acc0 = {0.f,0.f,0.f,0.f,0.f,0.f,0.f,0.f};
    v8f acc1 = acc0, acc2 = acc0, acc3 = acc0;
#pragma unroll
    for (int kc = 0; kc < 8; kc += 2) {
      const int offA0 = kc * 32 + lh * 16;
      const int offA1 = offA0 + 32;
      v16bf aE = ldfrag2(&zbuf[cur][am][offA0], &zbuf[cur][am][offA0 + 8]);
      v16bf aO = ldfrag2(&zbuf[cur][am][offA1], &zbuf[cur][am][offA1 + 8]);
      acc0 = __builtin_amdgcn_wmma_f32_16x16x32_bf16(false, aE, false, bh[kc],   (short)0, acc0, false, false);
      acc1 = __builtin_amdgcn_wmma_f32_16x16x32_bf16(false, aE, false, bl[kc],   (short)0, acc1, false, false);
      acc2 = __builtin_amdgcn_wmma_f32_16x16x32_bf16(false, aO, false, bh[kc+1], (short)0, acc2, false, false);
      acc3 = __builtin_amdgcn_wmma_f32_16x16x32_bf16(false, aO, false, bl[kc+1], (short)0, acc3, false, false);
    }
    v8f rcur = (acc0 + acc1) + (acc2 + acc3);
    v8f xin  = ldf8(xwp);                      // input currents, fragment layout

#pragma unroll
    for (int r = 0; r < 8; ++r) {
      int m = r + lh * 8;
      size_t idx = (size_t)(b0 + m) * N_SZ + n;
      float i_in  = xin[r] + rcur[r];
      float old_z = zreg[r];
      float na    = DECAY_C * areg[r] + (1.f - DECAY_C) * old_z;
      float thr   = THR_C + BETA_C * na;
      float nv    = DECAY_C * vreg[r] + (1.f - DECAY_C) * i_in - thr * old_z;
      float vsc   = (nv - thr) / thr;
      bool  spike = (vsc > 0.f) && (lreg[r] >= 2.0f);
      float z     = spike ? 1.f : 0.f;
      lreg[r] = (lreg[r] + 1.f) * (1.f - z);
      vreg[r] = nv; areg[r] = na; zreg[r] = z;
      zbuf[nxt][m][n] = spike ? (unsigned short)0x3F80 : (unsigned short)0; // bf16(1/0)

      size_t o = (size_t)t * plane + idx;      // out[4][T][B][N]
      out[o]             = z;
      out[tvar + o]      = nv;
      out[2 * tvar + o]  = thr;
      out[3 * tvar + o]  = vsc;
    }
  }
}

extern "C" void kernel_launch(void* const* d_in, const int* in_sizes, int n_in,
                              void* d_out, int out_size, void* d_ws, size_t ws_size,
                              hipStream_t stream) {
  const float* x     = (const float*)d_in[0];
  const float* w_in  = (const float*)d_in[1];
  const float* w_rec = (const float*)d_in[2];
  const float* z0    = (const float*)d_in[3];
  const float* v0    = (const float*)d_in[4];
  const float* a0    = (const float*)d_in[5];
  const float* lsd0  = (const float*)d_in[6];
  float* out = (float*)d_out;

  // workspace layout (all 16B-aligned): xwf | xh | xl | whT | wlT  (~131 MB)
  float*          xwf = (float*)d_ws;
  unsigned short* xh  = (unsigned short*)((char*)d_ws + (size_t)T_STEPS * B_SZ * N_SZ * 4);
  unsigned short* xl  = xh + NX_ELEMS;
  unsigned short* whT = xl + NX_ELEMS;
  unsigned short* wlT = whT + NW_ELEMS;

  lsnn_pack_x<<<8192, 256, 0, stream>>>(x, xh, xl);
  lsnn_pack_w<<<NW_ELEMS / 256, 256, 0, stream>>>(w_in, whT, wlT);
  // 500 t * 8 b-tiles * 16 n-tiles = 64000 wave-tiles, 4 waves/block
  lsnn_xw_gemm<<<16000, 128, 0, stream>>>(xh, xl, whT, wlT, xwf);
  // 8 independent batch-slices, 16 waves each
  lsnn_scan<<<8, 512, 0, stream>>>(w_rec, xwf, z0, v0, a0, lsd0, out);
}
